// MyMultiheadAttention_10015863734724
// MI455X (gfx1250) — compile-verified, matrix operand order checked by reading, not run on hardware
//
#include <hip/hip_runtime.h>
#include <cstdint>
#include <cstddef>

// ---------------- problem constants ----------------
constexpr int L_ = 2048;
constexpr int B_ = 2;
constexpr int E_ = 1024;
constexpr int H_ = 16;
constexpr int HD_ = 64;          // E/H
constexpr int M_ = L_ * B_;      // 4096 rows for the projection GEMMs

typedef __attribute__((ext_vector_type(16))) __bf16 v16bf;
typedef __attribute__((ext_vector_type(8)))  float  v8f;
typedef unsigned short ushort_t;

union FragBF {
    v16bf    v;
    uint32_t u[8];
};

__device__ __forceinline__ uint32_t f32_to_bf16_raw(float f) {
    union { float f; uint32_t u; } x{f};
    uint32_t r = x.u + 0x7FFFu + ((x.u >> 16) & 1u);   // round-to-nearest-even
    return r >> 16;
}
__device__ __forceinline__ uint32_t pack2(float lo, float hi) {
    return f32_to_bf16_raw(lo) | (f32_to_bf16_raw(hi) << 16);
}
__device__ __forceinline__ ushort_t bf16_scalar(float f) {
    return (ushort_t)f32_to_bf16_raw(f);
}

// Row reductions inside a 16-lane half-wave (rows of the C-layout live per half).
__device__ __forceinline__ float redmax16(float x) {
#pragma unroll
    for (int m = 1; m <= 8; m <<= 1) x = fmaxf(x, __shfl_xor(x, m, 32));
    return x;
}
__device__ __forceinline__ float redsum16(float x) {
#pragma unroll
    for (int m = 1; m <= 8; m <<= 1) x += __shfl_xor(x, m, 32);
    return x;
}

// K-offset within a 32-deep bf16 fragment for VGPR v and lane-half (ISA 7.12.2).
__device__ __forceinline__ int frag_koff(int v, int half) {
    return ((v & 4) ? 16 : 0) + half * 8 + (v & 3) * 2;
}

#define NEG_BIG (-3.0e38f)

// ---------- CDNA5 async-DMA helpers (gfx1250) ----------
typedef __attribute__((address_space(3))) uint32_t lds_u32_t;
__device__ __forceinline__ uint32_t lds_addr_of(const uint32_t* p) {
    // generic -> LDS addrspace -> 32-bit LDS byte offset
    return (uint32_t)(unsigned long long)(lds_u32_t*)(uint32_t*)p;
}
__device__ __forceinline__ void async_copy_b128(uint32_t lds_byte, const void* gsrc) {
    asm volatile("global_load_async_to_lds_b128 %0, %1, off"
                 :: "v"(lds_byte), "v"(gsrc) : "memory");
}
__device__ __forceinline__ void wait_asynccnt0() {
#if __has_builtin(__builtin_amdgcn_s_wait_asynccnt)
    __builtin_amdgcn_s_wait_asynccnt(0);
#else
    asm volatile("s_wait_asynccnt 0" ::: "memory");
#endif
}

// =====================================================================
// f32 -> bf16 bulk convert (pairs). Memory-bound one-shot pass so the
// GEMM/attention hot loops never touch f32->bf16 VALU math again.
// =====================================================================
__global__ __launch_bounds__(256) void cvt_f32_bf16(
    const float* __restrict__ in, uint32_t* __restrict__ out, int npairs)
{
    int i      = blockIdx.x * 256 + threadIdx.x;
    int stride = gridDim.x * 256;
    for (; i < npairs; i += stride) {
        float2 v = ((const float2*)in)[i];
        out[i] = pack2(v.x, v.y);
    }
}

// =====================================================================
// GEMM: C[M,N] = A[M,K] @ W[N,K]^T + bias[N]   (bf16 in, f32 accum)
// 128x128 tile, K-step 32, 8 waves (4m x 2n), 2x4 WMMA tiles per wave.
// LDS double-buffered; tiles staged with global_load_async_to_lds_b128
// (ASYNCcnt) so DMA of tile k+1 overlaps WMMA on tile k.
// =====================================================================
template <bool OUT_BF16>
__global__ __launch_bounds__(256) void gemm_bf16_wmma(
    const ushort_t* __restrict__ A, const ushort_t* __restrict__ W,
    const float* __restrict__ bias, float* __restrict__ Cf,
    ushort_t* __restrict__ Cb, int M, int N, int K)
{
    // [buf][operand][128 rows x 16 dword-pairs] = 32 KB total
    __shared__ uint32_t smem[2][2][128 * 16];

    const int tid  = threadIdx.x;
    const int lane = tid & 31;
    const int wid  = tid >> 5;
    const int half = lane >> 4;
    const int nl   = lane & 15;

    const int m0 = blockIdx.y * 128;
    const int n0 = blockIdx.x * 128;
    const int wm = (wid & 3) * 32;
    const int wn = (wid >> 2) * 64;

    v8f acc[2][4];
#pragma unroll
    for (int i = 0; i < 2; ++i)
#pragma unroll
        for (int j = 0; j < 4; ++j) acc[i][j] = v8f{};

    // Stage one 128x32 bf16 tile (8 KB): 512 x b128, 2 per thread.
    auto stage = [&](int buf, int op, const ushort_t* g, int row0, int kk) {
        uint32_t base = lds_addr_of(&smem[buf][op][0]);
#pragma unroll
        for (int i = 0; i < 2; ++i) {
            int f  = tid + i * 256;       // 0..511
            int r  = f >> 2;              // 0..127
            int cb = (f & 3) * 16;        // byte offset within 64B row
            const void* src = (const char*)(g + (size_t)(row0 + r) * K + kk) + cb;
            async_copy_b128(base + (uint32_t)(r * 64 + cb), src);
        }
    };

    stage(0, 0, A, m0, 0);
    stage(0, 1, W, n0, 0);

    int cur = 0;
    for (int kk = 0; kk < K; kk += 32) {
        wait_asynccnt0();
        __syncthreads();
        if (kk + 32 < K) {
            stage(cur ^ 1, 0, A, m0, kk + 32);
            stage(cur ^ 1, 1, W, n0, kk + 32);
        }

        const uint32_t* Ap = &smem[cur][0][0];
        const uint32_t* Bp = &smem[cur][1][0];

        FragBF af[2], bf[4];
#pragma unroll
        for (int i = 0; i < 2; ++i) {
            int row = wm + i * 16 + nl;
#pragma unroll
            for (int v = 0; v < 8; ++v)
                af[i].u[v] = Ap[row * 16 + (frag_koff(v, half) >> 1)];
        }
#pragma unroll
        for (int j = 0; j < 4; ++j) {
            int row = wn + j * 16 + nl;
#pragma unroll
            for (int v = 0; v < 8; ++v)
                bf[j].u[v] = Bp[row * 16 + (frag_koff(v, half) >> 1)];
        }
#pragma unroll
        for (int i = 0; i < 2; ++i)
#pragma unroll
            for (int j = 0; j < 4; ++j)
                acc[i][j] = __builtin_amdgcn_wmma_f32_16x16x32_bf16(
                    false, af[i].v, false, bf[j].v, (short)0, acc[i][j], false, false);
        cur ^= 1;
    }

    // epilogue: C layout -> lane n = lane&15, row m = v + half*8
#pragma unroll
    for (int j = 0; j < 4; ++j) {
        int ncol = n0 + wn + j * 16 + nl;
        float bv = bias[ncol];
#pragma unroll
        for (int i = 0; i < 2; ++i) {
#pragma unroll
            for (int v = 0; v < 8; ++v) {
                int mrow  = m0 + wm + i * 16 + v + half * 8;
                float val = acc[i][j][v] + bv;
                if constexpr (OUT_BF16)
                    Cb[(size_t)mrow * N + ncol] = bf16_scalar(val);
                else
                    Cf[(size_t)mrow * N + ncol] = val;
            }
        }
    }
}

// =====================================================================
// Flash-attention forward (causal), bf16 Q/K/V. One wave = 16 query
// rows; barrier-free (per-wave causal trip count). Scale applied to the
// f32 score accumulator. Writes bf16 O and per-row (max, sumexp) stats.
// =====================================================================
__global__ __launch_bounds__(256) void flash_attn_fwd(
    const ushort_t* __restrict__ Qb, const ushort_t* __restrict__ Kb,
    const ushort_t* __restrict__ Vb, ushort_t* __restrict__ Ob,
    float* __restrict__ stats)
{
    __shared__ __align__(4) ushort_t pbuf[8][16 * 32]; // wave-private P tiles

    const int tid  = threadIdx.x;
    const int lane = tid & 31;
    const int wid  = tid >> 5;
    const int half = lane >> 4;
    const int nl   = lane & 15;

    const int b  = blockIdx.y / H_;
    const int h  = blockIdx.y % H_;
    const int q0 = blockIdx.x * 128 + wid * 16;
    const float scale = 0.125f;                 // 1/sqrt(64)

    // Resident Q fragments (A layout: lane m = lane&15) — raw dword pairs
    FragBF qf[2];
    {
        const ushort_t* qrow = Qb + ((size_t)(q0 + nl) * B_ + b) * E_ + h * HD_;
#pragma unroll
        for (int i = 0; i < 2; ++i)
#pragma unroll
            for (int v = 0; v < 8; ++v)
                qf[i].u[v] = *(const uint32_t*)(qrow + i * 32 + frag_koff(v, half));
    }

    v8f oacc[4];
#pragma unroll
    for (int dt = 0; dt < 4; ++dt) oacc[dt] = v8f{};
    float rm[8], rl[8];
#pragma unroll
    for (int v = 0; v < 8; ++v) { rm[v] = NEG_BIG; rl[v] = 0.f; }

    const int jend = q0 + 16;                    // causal: keys <= q0+15
    for (int j0 = 0; j0 < jend; j0 += 32) {
        // ---- scores S[16x32] = Q @ K^T ----
        v8f sacc[2] = { v8f{}, v8f{} };
#pragma unroll
        for (int t = 0; t < 2; ++t) {
            FragBF kf0, kf1;
            const ushort_t* krp = Kb + ((size_t)(j0 + t * 16 + nl) * B_ + b) * E_ + h * HD_;
#pragma unroll
            for (int v = 0; v < 8; ++v) {
                int c = frag_koff(v, half);
                kf0.u[v] = *(const uint32_t*)(krp + c);
                kf1.u[v] = *(const uint32_t*)(krp + 32 + c);
            }
            sacc[t] = __builtin_amdgcn_wmma_f32_16x16x32_bf16(
                false, qf[0].v, false, kf0.v, (short)0, sacc[t], false, false);
            sacc[t] = __builtin_amdgcn_wmma_f32_16x16x32_bf16(
                false, qf[1].v, false, kf1.v, (short)0, sacc[t], false, false);
        }

        // ---- scale + causal mask + online softmax ----
#pragma unroll
        for (int v = 0; v < 8; ++v) {
            int qi = q0 + v + half * 8;
            float s0 = (j0 + nl      <= qi) ? sacc[0][v] * scale : NEG_BIG;
            float s1 = (j0 + 16 + nl <= qi) ? sacc[1][v] * scale : NEG_BIG;
            float mx = redmax16(fmaxf(s0, s1));
            float mn = fmaxf(rm[v], mx);
            float alpha = __expf(rm[v] - mn);
            float p0 = __expf(s0 - mn);
            float p1 = __expf(s1 - mn);
            rl[v] = rl[v] * alpha + redsum16(p0 + p1);
            rm[v] = mn;
#pragma unroll
            for (int dt = 0; dt < 4; ++dt) oacc[dt][v] *= alpha;
            int row = v + half * 8;
            pbuf[wid][row * 32 + nl]      = bf16_scalar(p0);
            pbuf[wid][row * 32 + 16 + nl] = bf16_scalar(p1);
        }

        asm volatile("s_wait_dscnt 0" ::: "memory");

        // reload P as A fragment
        FragBF pf;
        const uint32_t* pp = (const uint32_t*)&pbuf[wid][0];
#pragma unroll
        for (int v = 0; v < 8; ++v)
            pf.u[v] = pp[nl * 16 + (frag_koff(v, half) >> 1)];

        // ---- O += P @ V ----
#pragma unroll
        for (int dt = 0; dt < 4; ++dt) {
            FragBF vf;
#pragma unroll
            for (int v = 0; v < 8; ++v) {
                int c = frag_koff(v, half);    // key offset (contract dim)
                const ushort_t* vp = Vb + ((size_t)(j0 + c) * B_ + b) * E_
                                        + h * HD_ + dt * 16 + nl;
                vf.u[v] = (uint32_t)vp[0] | ((uint32_t)vp[(size_t)B_ * E_] << 16);
            }
            oacc[dt] = __builtin_amdgcn_wmma_f32_16x16x32_bf16(
                false, pf.v, false, vf.v, (short)0, oacc[dt], false, false);
        }
    }

    // epilogue: normalize + store bf16 O and f32 stats
#pragma unroll
    for (int v = 0; v < 8; ++v) {
        int l = q0 + v + half * 8;
        float inv = 1.0f / rl[v];
        ushort_t* orow = Ob + ((size_t)l * B_ + b) * E_ + h * HD_;
#pragma unroll
        for (int dt = 0; dt < 4; ++dt)
            orow[dt * 16 + nl] = bf16_scalar(oacc[dt][v] * inv);
        if (nl == 0) {
            float* st = stats + (((size_t)b * H_ + h) * L_ + l) * 2;
            st[0] = rm[v];
            st[1] = rl[v];
        }
    }
}

// =====================================================================
// attn_avg[b,q,j] = mean_h softmax(QK^T)[b,h,q,j] — recompute scores via
// WMMA using saved stats. Each wave: 16q x 32j tile, loops 16 heads.
// Zeros written above the diagonal (d_out is poisoned).
// =====================================================================
__global__ __launch_bounds__(256) void attn_avg_kernel(
    const ushort_t* __restrict__ Qb, const ushort_t* __restrict__ Kb,
    const float* __restrict__ stats, float* __restrict__ attn)
{
    const int tid  = threadIdx.x;
    const int lane = tid & 31;
    const int wid  = tid >> 5;
    const int half = lane >> 4;
    const int nl   = lane & 15;

    const int j0 = blockIdx.x * 32;
    const int q0 = blockIdx.y * 128 + wid * 16;
    const int b  = blockIdx.z;
    const float scale = 0.125f;

    float acc0[8] = {0.f,0.f,0.f,0.f,0.f,0.f,0.f,0.f};
    float acc1[8] = {0.f,0.f,0.f,0.f,0.f,0.f,0.f,0.f};

    if (j0 <= q0 + 15) {
        for (int h = 0; h < H_; ++h) {
            FragBF qf[2];
            const ushort_t* qrow = Qb + ((size_t)(q0 + nl) * B_ + b) * E_ + h * HD_;
#pragma unroll
            for (int i = 0; i < 2; ++i)
#pragma unroll
                for (int v = 0; v < 8; ++v)
                    qf[i].u[v] = *(const uint32_t*)(qrow + i * 32 + frag_koff(v, half));

            v8f sacc[2] = { v8f{}, v8f{} };
#pragma unroll
            for (int t = 0; t < 2; ++t) {
                FragBF kf0, kf1;
                const ushort_t* krp = Kb + ((size_t)(j0 + t * 16 + nl) * B_ + b) * E_ + h * HD_;
#pragma unroll
                for (int v = 0; v < 8; ++v) {
                    int c = frag_koff(v, half);
                    kf0.u[v] = *(const uint32_t*)(krp + c);
                    kf1.u[v] = *(const uint32_t*)(krp + 32 + c);
                }
                sacc[t] = __builtin_amdgcn_wmma_f32_16x16x32_bf16(
                    false, qf[0].v, false, kf0.v, (short)0, sacc[t], false, false);
                sacc[t] = __builtin_amdgcn_wmma_f32_16x16x32_bf16(
                    false, qf[1].v, false, kf1.v, (short)0, sacc[t], false, false);
            }
#pragma unroll
            for (int v = 0; v < 8; ++v) {
                int qi = q0 + v + half * 8;
                const float* st = stats + (((size_t)b * H_ + h) * L_ + qi) * 2;
                float mrow = st[0];
                float inv  = 1.0f / st[1];
                if (j0 + nl <= qi)
                    acc0[v] += __expf(sacc[0][v] * scale - mrow) * inv;
                if (j0 + 16 + nl <= qi)
                    acc1[v] += __expf(sacc[1][v] * scale - mrow) * inv;
            }
        }
    }

    const float invH = 1.0f / (float)H_;
#pragma unroll
    for (int v = 0; v < 8; ++v) {
        int qi = q0 + v + half * 8;
        float* row = attn + ((size_t)b * L_ + qi) * L_ + j0;
        row[nl]      = acc0[v] * invH;
        row[16 + nl] = acc1[v] * invH;
    }
}

// =====================================================================
extern "C" void kernel_launch(void* const* d_in, const int* in_sizes, int n_in,
                              void* d_out, int out_size, void* d_ws, size_t ws_size,
                              hipStream_t stream)
{
    (void)in_sizes; (void)n_in; (void)out_size; (void)ws_size;
    const float* query = (const float*)d_in[0];
    const float* key   = (const float*)d_in[1];
    const float* value = (const float*)d_in[2];
    const float* w_in  = (const float*)d_in[3];   // [3E, E]
    const float* b_in  = (const float*)d_in[4];   // [3E]
    const float* w_out = (const float*)d_in[5];   // [E, E]
    const float* b_out = (const float*)d_in[6];   // [E]

    float* out  = (float*)d_out;                  // [L,B,E]
    float* attn = out + (size_t)L_ * B_ * E_;     // [B,L,L]

    // workspace layout (bf16 stored as ushort; everything 4B aligned)
    ushort_t* qbf   = (ushort_t*)d_ws;                 // [M,E] projected Q
    ushort_t* kbf   = qbf   + (size_t)M_ * E_;         // [M,E] projected K
    ushort_t* vbf   = kbf   + (size_t)M_ * E_;         // [M,E] projected V
    ushort_t* obf   = vbf   + (size_t)M_ * E_;         // [M,E] attn output
    ushort_t* xq    = obf   + (size_t)M_ * E_;         // [M,E] query (bf16)
    ushort_t* xk    = xq    + (size_t)M_ * E_;
    ushort_t* xv    = xk    + (size_t)M_ * E_;
    ushort_t* wqkvb = xv    + (size_t)M_ * E_;         // [3E,E] bf16 weights
    ushort_t* woutb = wqkvb + (size_t)3 * E_ * E_;     // [E,E]
    float*    stats = (float*)(woutb + (size_t)E_ * E_); // [B,H,L,2]

    // one-shot f32 -> bf16 conversions (memory-bound)
    const int act_pairs = (M_ * E_) / 2;               // 2M pairs
    cvt_f32_bf16<<<1024, 256, 0, stream>>>(query, (uint32_t*)xq, act_pairs);
    cvt_f32_bf16<<<1024, 256, 0, stream>>>(key,   (uint32_t*)xk, act_pairs);
    cvt_f32_bf16<<<1024, 256, 0, stream>>>(value, (uint32_t*)xv, act_pairs);
    cvt_f32_bf16<<<1024, 256, 0, stream>>>(w_in,  (uint32_t*)wqkvb, (3 * E_ * E_) / 2);
    cvt_f32_bf16<<<1024, 256, 0, stream>>>(w_out, (uint32_t*)woutb, (E_ * E_) / 2);

    dim3 gp(E_ / 128, M_ / 128);                       // (8, 32)
    gemm_bf16_wmma<true><<<gp, 256, 0, stream>>>(
        xq, wqkvb,                        b_in,          nullptr, qbf, M_, E_, E_);
    gemm_bf16_wmma<true><<<gp, 256, 0, stream>>>(
        xk, wqkvb + (size_t)E_ * E_,      b_in + E_,     nullptr, kbf, M_, E_, E_);
    gemm_bf16_wmma<true><<<gp, 256, 0, stream>>>(
        xv, wqkvb + (size_t)2 * E_ * E_,  b_in + 2 * E_, nullptr, vbf, M_, E_, E_);

    flash_attn_fwd<<<dim3(L_ / 128, B_ * H_), 256, 0, stream>>>(qbf, kbf, vbf, obf, stats);

    attn_avg_kernel<<<dim3(L_ / 32, L_ / 128, B_), 256, 0, stream>>>(qbf, kbf, stats, attn);

    gemm_bf16_wmma<false><<<gp, 256, 0, stream>>>(
        obf, woutb, b_out, out, nullptr, M_, E_, E_);
}
